// MultiHeadAttention_45561013076163
// MI455X (gfx1250) — matrix operand order repaired and checked by reading, not compiled, not run on hardware
//
#include <hip/hip_runtime.h>
#include <hip/hip_bf16.h>

typedef __attribute__((ext_vector_type(16))) _Float16 v16h;
typedef __attribute__((ext_vector_type(8)))  float    v8f;

#define NEG_BIG (-1.0e9f)

union HU { v16h h; uint4 q[2]; };

// A-fragment (16x32, f16) per CDNA5 ISA 7.12.2:
// lane r (0..15), half hf: M=r, elems 0..7 -> K = hf*8 + 0..7, elems 8..15 -> K = 16 + hf*8 + 0..7
__device__ __forceinline__ v16h load_a16(const _Float16* base, int lda) {
  const int lane = threadIdx.x & 31;
  const int r = lane & 15, hf = lane >> 4;
  const _Float16* p = base + (size_t)r * lda + hf * 8;
  HU u;
  u.q[0] = *reinterpret_cast<const uint4*>(p);
  u.q[1] = *reinterpret_cast<const uint4*>(p + 16);
  return u.h;
}

// B-fragment (32x16, f16): lane r holds column N=r, half hf: elems 0..7 -> K = hf*8 + 0..7, elems 8..15 -> K = 16 + hf*8 + 0..7.
// Source matrix is stored [N, K] row-major (K contiguous), leading dim ldb.
__device__ __forceinline__ v16h load_b16(const _Float16* base, int ldb) {
  const int lane = threadIdx.x & 31;
  const int r = lane & 15, hf = lane >> 4;
  const _Float16* p = base + (size_t)r * ldb + hf * 8;
  HU u;
  u.q[0] = *reinterpret_cast<const uint4*>(p);
  u.q[1] = *reinterpret_cast<const uint4*>(p + 16);
  return u.h;
}

__device__ __forceinline__ v8f wmma_f16(v16h a, v16h b, v8f c) {
  return __builtin_amdgcn_wmma_f32_16x16x32_f16(false, a, false, b, (short)0, c, false, false);
}

// Async copy 16B/lane from global to LDS (CDNA5 GLOBAL_LOAD_ASYNC_TO_LDS_B128, ASYNCcnt).
// LDS byte address = low 32 bits of the flat shared-aperture address (ISA 10.2).
__device__ __forceinline__ void async_g2l_b128(const void* gp, const void* lp) {
  const unsigned long long ga = (unsigned long long)(uintptr_t)gp;
  const unsigned int loff = (unsigned int)(uintptr_t)lp;
  asm volatile("global_load_async_to_lds_b128 %0, %1, off" :: "v"(loff), "v"(ga) : "memory");
}
__device__ __forceinline__ void wait_async0() {
  asm volatile("s_wait_asynccnt 0x0" ::: "memory");
}

__global__ void cvt_f32_f16(const float* __restrict__ in, _Float16* __restrict__ out, int n) {
  int i = blockIdx.x * blockDim.x + threadIdx.x;
  if (i < n) out[i] = (_Float16)in[i];
}

// C = A(MxK, row-major) * Bt(NxK, row-major)^T.
// Block = 4 waves; block tile = 64(M) x 64(N); each wave owns 16 M-rows.
// Per 32-wide K step the block async-stages the shared B slice (64x32 f16) into LDS
// (rows padded to 40 halves for bank-conflict-free b128 fragment reads), giving 4x reuse.
// MODE 0: f16 row-major out; MODE 1: f16 transposed out ([B, N, batchS] with m=(b,s));
// MODE 2: f32 row-major out.
template <int MODE>
__global__ __launch_bounds__(128) void gemm_nt(
    const _Float16* __restrict__ A, const _Float16* __restrict__ Bt,
    void* __restrict__ Out, int M, int N, int K, int batchS) {
  __shared__ __align__(16) _Float16 sB[64][40];   // 64 N-rows x 32 K (padded to 40)
  const int tid = threadIdx.x;
  const int w = tid >> 5;
  const int row0 = blockIdx.x * 64 + w * 16;
  const int col0 = blockIdx.y * 64;

  v8f acc[4] = {};
  for (int k0 = 0; k0 < K; k0 += 32) {
    // cooperative async stage: 64 rows x 64B = 256 b128 transfers, 2 per thread
#pragma unroll
    for (int j = 0; j < 2; ++j) {
      const int f = j * 128 + tid;          // 0..255
      const int rowB = f >> 2, seg = f & 3; // 4 x 16B segments per row
      async_g2l_b128(Bt + (size_t)(col0 + rowB) * K + k0 + seg * 8,
                     &sB[rowB][seg * 8]);
    }
    wait_async0();
    __syncthreads();

    v16h a = load_a16(A + (size_t)row0 * K + k0, K);
#pragma unroll
    for (int t = 0; t < 4; ++t) {
      v16h b = load_b16(&sB[t * 16][0], 40);
      acc[t] = wmma_f16(a, b, acc[t]);
    }
    __syncthreads();   // all fragment reads done before next stage overwrites sB
  }

  const int lane = tid & 31, r = lane & 15, hf = lane >> 4;
#pragma unroll
  for (int t = 0; t < 4; ++t) {
#pragma unroll
    for (int i = 0; i < 8; ++i) {
      const int m = row0 + hf * 8 + i;     // C/D layout: VGPR i -> M = 8*hf + i, lane r -> N
      const int n = col0 + t * 16 + r;
      const float v = acc[t][i];
      if (MODE == 0) {
        ((_Float16*)Out)[(size_t)m * N + n] = (_Float16)v;
      } else if (MODE == 1) {
        const int bb = m / batchS, s = m - bb * batchS;
        ((_Float16*)Out)[((size_t)bb * N + n) * batchS + s] = (_Float16)v;
      } else {
        ((float*)Out)[(size_t)m * N + n] = v;
      }
    }
  }
}

// Flash attention: one wave handles 16 queries of one (b, h); loops over keys in blocks of 32.
// Q, Kh: [B, S, D] f16 (head h = cols h*DK .. h*DK+DK-1). Vt: [B, D, S] f16 (key dim contiguous).
__global__ __launch_bounds__(32) void flash_attn(
    const _Float16* __restrict__ Q, const _Float16* __restrict__ Kh,
    const _Float16* __restrict__ Vt, const int* __restrict__ valid_lens,
    _Float16* __restrict__ Ctx, int S, int D, int DK) {
  const int qb = blockIdx.x, h = blockIdx.y, b = blockIdx.z;
  const int lane = threadIdx.x & 31, r = lane & 15, hf = lane >> 4;
  __shared__ __align__(16) _Float16 sP[16][32];

  const int q0 = qb * 16;
  const _Float16* Qp = Q + (size_t)b * S * D + h * DK;
  const _Float16* Kp = Kh + (size_t)b * S * D + h * DK;
  const _Float16* Vp = Vt + ((size_t)b * D + h * DK) * S;
  const int vl = valid_lens[b];

  // Q block fragments: 16 queries x 64 head-dims = two K=32 fragments
  const v16h qa0 = load_a16(Qp + (size_t)q0 * D, D);
  const v16h qa1 = load_a16(Qp + (size_t)q0 * D + 32, D);

  v8f o[4] = {};
  float rmax[8], rsum[8];
#pragma unroll
  for (int i = 0; i < 8; ++i) { rmax[i] = -INFINITY; rsum[i] = 0.0f; }

  for (int kb = 0; kb < S; kb += 32) {
    // scores for two 16-key tiles
    v8f s0 = {}, s1 = {};
    {
      v16h b0  = load_b16(Kp + (size_t)kb * D, D);
      s0 = wmma_f16(qa0, b0, s0);
      v16h b0h = load_b16(Kp + (size_t)kb * D + 32, D);
      s0 = wmma_f16(qa1, b0h, s0);
      v16h b1  = load_b16(Kp + (size_t)(kb + 16) * D, D);
      s1 = wmma_f16(qa0, b1, s1);
      v16h b1h = load_b16(Kp + (size_t)(kb + 16) * D + 32, D);
      s1 = wmma_f16(qa1, b1h, s1);
    }
    const bool msk0 = (kb + r) >= vl;
    const bool msk1 = (kb + 16 + r) >= vl;

    float p0[8], p1[8], alpha[8];
#pragma unroll
    for (int i = 0; i < 8; ++i) {
      // row m = 8*hf + i; its 16 values per tile live in VGPR i across one 16-lane group
      float x0 = msk0 ? NEG_BIG : s0[i] * 0.125f;   // 1/sqrt(64)
      float x1 = msk1 ? NEG_BIG : s1[i] * 0.125f;
      float t = fmaxf(x0, x1);
      t = fmaxf(t, __shfl_xor(t, 1));
      t = fmaxf(t, __shfl_xor(t, 2));
      t = fmaxf(t, __shfl_xor(t, 4));
      t = fmaxf(t, __shfl_xor(t, 8));
      const float nmax = fmaxf(rmax[i], t);
      p0[i] = __expf(x0 - nmax);
      p1[i] = __expf(x1 - nmax);
      float ps = p0[i] + p1[i];
      ps += __shfl_xor(ps, 1);
      ps += __shfl_xor(ps, 2);
      ps += __shfl_xor(ps, 4);
      ps += __shfl_xor(ps, 8);
      alpha[i] = __expf(rmax[i] - nmax);
      rsum[i] = rsum[i] * alpha[i] + ps;
      rmax[i] = nmax;
    }

    // stage P (C/D layout) through LDS into A-fragment layout
    __syncthreads();
#pragma unroll
    for (int i = 0; i < 8; ++i) {
      sP[hf * 8 + i][r]      = (_Float16)p0[i];
      sP[hf * 8 + i][16 + r] = (_Float16)p1[i];
    }
    __syncthreads();
    const v16h pa = load_a16(&sP[0][0], 32);

#pragma unroll
    for (int t4 = 0; t4 < 4; ++t4) {
#pragma unroll
      for (int i = 0; i < 8; ++i) o[t4][i] *= alpha[i];
      v16h vb = load_b16(Vp + (size_t)(t4 * 16) * S + kb, S);
      o[t4] = wmma_f16(pa, vb, o[t4]);
    }
  }

  float inv[8];
#pragma unroll
  for (int i = 0; i < 8; ++i) inv[i] = 1.0f / rsum[i];
  _Float16* Cp = Ctx + (size_t)b * S * D + h * DK;
#pragma unroll
  for (int t4 = 0; t4 < 4; ++t4) {
#pragma unroll
    for (int i = 0; i < 8; ++i) {
      Cp[(size_t)(q0 + hf * 8 + i) * D + t4 * 16 + r] = (_Float16)(o[t4][i] * inv[i]);
    }
  }
}

extern "C" void kernel_launch(void* const* d_in, const int* in_sizes, int n_in,
                              void* d_out, int out_size, void* d_ws, size_t ws_size,
                              hipStream_t stream) {
  const int B = 2, S = 2048, D = 1024, H = 16, DK = 64;
  const float* X  = (const float*)d_in[0];
  const int* vlens = (const int*)d_in[1];
  const float* Wq = (const float*)d_in[2];
  const float* Wk = (const float*)d_in[3];
  const float* Wv = (const float*)d_in[4];
  const float* Wo = (const float*)d_in[5];

  const size_t szX = (size_t)B * S * D;  // 4,194,304
  const size_t szW = (size_t)D * D;      // 1,048,576

  _Float16* ws  = (_Float16*)d_ws;
  _Float16* Xh  = ws;
  _Float16* Wqh = Xh + szX;
  _Float16* Wkh = Wqh + szW;
  _Float16* Wvh = Wkh + szW;
  _Float16* Woh = Wvh + szW;
  _Float16* Qb  = Woh + szW;
  _Float16* Kb  = Qb + szX;
  _Float16* Vt  = Kb + szX;   // [B, D, S] transposed V
  _Float16* Ctx = Vt + szX;

  const int thr = 256;
  cvt_f32_f16<<<(int)((szX + thr - 1) / thr), thr, 0, stream>>>(X,  Xh,  (int)szX);
  cvt_f32_f16<<<(int)((szW + thr - 1) / thr), thr, 0, stream>>>(Wq, Wqh, (int)szW);
  cvt_f32_f16<<<(int)((szW + thr - 1) / thr), thr, 0, stream>>>(Wk, Wkh, (int)szW);
  cvt_f32_f16<<<(int)((szW + thr - 1) / thr), thr, 0, stream>>>(Wv, Wvh, (int)szW);
  cvt_f32_f16<<<(int)((szW + thr - 1) / thr), thr, 0, stream>>>(Wo, Woh, (int)szW);

  dim3 blk(128);
  dim3 g((B * S) / 64, D / 64);
  gemm_nt<0><<<g, blk, 0, stream>>>(Xh, Wqh, (void*)Qb, B * S, D, D, S);
  gemm_nt<0><<<g, blk, 0, stream>>>(Xh, Wkh, (void*)Kb, B * S, D, D, S);
  gemm_nt<1><<<g, blk, 0, stream>>>(Xh, Wvh, (void*)Vt, B * S, D, D, S);

  dim3 gf(S / 16, H, B);
  flash_attn<<<gf, dim3(32), 0, stream>>>(Qb, Kb, Vt, vlens, Ctx, S, D, DK);

  gemm_nt<2><<<g, blk, 0, stream>>>(Ctx, Woh, d_out, B * S, D, D, S);
}
